// SelfAttention_47004122087726
// MI455X (gfx1250) — matrix operand order repaired and checked by reading, not compiled, not run on hardware
//
#include <hip/hip_runtime.h>
#include <hip/hip_bf16.h>

typedef __bf16 bf16_t;
typedef __attribute__((ext_vector_type(16))) __bf16 v16bf;
typedef __attribute__((ext_vector_type(8)))  float  v8f;

constexpr int N = 8, L = 1024, E = 1024, H = 16, D = 64;

union BF16x16 { v16bf v; uint4 u[2]; };

__device__ __forceinline__ v8f wmma_bf16(v16bf a, v16bf b, v8f c) {
  // D = A(16x32 bf16) * B(32x16 bf16) + C(16x16 f32)
  return __builtin_amdgcn_wmma_f32_16x16x32_bf16(
      false, a, false, b, (short)0, c, false, false);
}

// A-fragment (16x32) from row-major bf16 matrix, rows m0..m0+15, k in [kbase,kbase+32)
__device__ __forceinline__ v16bf load_a_bf16(const bf16_t* p0, int ld, int m0,
                                             int kbase, int lane) {
  const int hf = lane >> 4;
  const bf16_t* p = p0 + (size_t)(m0 + (lane & 15)) * ld + kbase + hf * 8;
  BF16x16 f;
  f.u[0] = *reinterpret_cast<const uint4*>(p);        // K: +0..7  (half 0) / +8..15 (half 1)
  f.u[1] = *reinterpret_cast<const uint4*>(p + 16);   // K: +16..23 / +24..31
  return f.v;
}

// B-fragment (32x16) from k-major bf16 matrix: element (k,n) at p0[k*ld + n]
__device__ __forceinline__ v16bf load_b_bf16(const bf16_t* p0, int ld, int kbase,
                                             int n0, int lane) {
  const int hf = lane >> 4;
  const bf16_t* p = p0 + (size_t)(kbase + hf * 8) * ld + n0 + (lane & 15);
  v16bf v;
#pragma unroll
  for (int i = 0; i < 8; ++i) v[i] = p[(size_t)i * ld];
#pragma unroll
  for (int i = 0; i < 8; ++i) v[8 + i] = p[(size_t)(16 + i) * ld];
  return v;
}

// B-fragment from f32 source with convert (weights)
__device__ __forceinline__ v16bf load_b_f32cvt(const float* p0, int ld, int kbase,
                                               int n0, int lane) {
  const int hf = lane >> 4;
  const float* p = p0 + (size_t)(kbase + hf * 8) * ld + n0 + (lane & 15);
  v16bf v;
#pragma unroll
  for (int i = 0; i < 8; ++i) v[i] = (__bf16)p[(size_t)i * ld];
#pragma unroll
  for (int i = 0; i < 8; ++i) v[8 + i] = (__bf16)p[(size_t)(16 + i) * ld];
  return v;
}

// A-fragment from f32 source with convert (x rows / LDS f32 tiles)
__device__ __forceinline__ v16bf load_a_f32cvt(const float* p0, int ld, int m0,
                                               int kbase, int lane) {
  const int hf = lane >> 4;
  const float* p = p0 + (size_t)(m0 + (lane & 15)) * ld + kbase + hf * 8;
  v16bf v;
#pragma unroll
  for (int i = 0; i < 8; ++i) v[i] = (__bf16)p[i];
#pragma unroll
  for (int i = 0; i < 8; ++i) v[8 + i] = (__bf16)p[16 + i];
  return v;
}

// ---------------- Kernel 1: QKV projection ----------------
// x viewed as (N*L*H, 64) row-major (rows contiguous). One wave -> 16 rows.
__global__ __launch_bounds__(32) void proj_kernel(
    const float* __restrict__ x,
    const float* __restrict__ Wq, const float* __restrict__ Wk,
    const float* __restrict__ Wv,
    bf16_t* __restrict__ Qs,   // (N,H,L,D) ; pre-scaled by 0.06
    bf16_t* __restrict__ Kt,   // (N,H,D,L) transposed
    bf16_t* __restrict__ Vs)   // (N,H,L,D)
{
  const int lane = threadIdx.x;
  const int hf = lane >> 4;
  const int nm = lane & 15;
  const int row0 = blockIdx.x * 16;

  v16bf a0 = load_a_f32cvt(x, D, row0, 0, lane);
  v16bf a1 = load_a_f32cvt(x, D, row0, 32, lane);

  const float* Ws[3] = {Wq, Wk, Wv};
#pragma unroll
  for (int w = 0; w < 3; ++w) {
#pragma unroll
    for (int dt = 0; dt < 4; ++dt) {
      v8f c = {};
      c = wmma_bf16(a0, load_b_f32cvt(Ws[w], D, 0, dt * 16, lane), c);
      c = wmma_bf16(a1, load_b_f32cvt(Ws[w], D, 32, dt * 16, lane), c);
#pragma unroll
      for (int r = 0; r < 8; ++r) {
        const int M = r + hf * 8;
        const int R = row0 + M;           // R = (n*L + l)*H + h
        const int h = R % H;
        const int l = (R / H) % L;
        const int n = R / (H * L);
        const int d = dt * 16 + nm;
        const size_t hb = (size_t)(n * H + h);
        if (w == 0)      Qs[(hb * L + l) * D + d] = (__bf16)(c[r] * 0.06f);
        else if (w == 1) Kt[(hb * D + d) * L + l] = (__bf16)c[r];
        else             Vs[(hb * L + l) * D + d] = (__bf16)c[r];
      }
    }
  }
}

// ---------------- Kernel 2: attention + fc ----------------
// One wave handles 16 query rows of one (n,h): scores -> softmax -> PV -> fc.
__global__ __launch_bounds__(32) void attn_kernel(
    const bf16_t* __restrict__ Qs, const bf16_t* __restrict__ Kt,
    const bf16_t* __restrict__ Vs, const float* __restrict__ Wfc,
    float* __restrict__ out, float* __restrict__ attn)
{
  __shared__ __align__(16) float S[16 * L];     // 64 KB score/prob tile
  __shared__ __align__(16) float Obuf[16 * D];  // 4 KB  O tile

  const int lane = threadIdx.x;
  const int hf = lane >> 4;
  const int nm = lane & 15;
  const int m0 = blockIdx.x * 16;
  const int h = blockIdx.y;
  const int n = blockIdx.z;

  const size_t hb = (size_t)(n * H + h);
  const bf16_t* Qh = Qs + hb * (size_t)L * D;
  const bf16_t* Kh = Kt + hb * (size_t)D * L;
  const bf16_t* Vh = Vs + hb * (size_t)L * D;

  v16bf aq0 = load_a_bf16(Qh, D, m0, 0, lane);
  v16bf aq1 = load_a_bf16(Qh, D, m0, 32, lane);

  // S = (Q*0.06) K^T / sqrt(E)
  for (int kb = 0; kb < L / 16; ++kb) {
    v8f c = {};
    c = wmma_bf16(aq0, load_b_bf16(Kh, L, 0, kb * 16, lane), c);
    c = wmma_bf16(aq1, load_b_bf16(Kh, L, 32, kb * 16, lane), c);
#pragma unroll
    for (int r = 0; r < 8; ++r)
      S[(r + hf * 8) * L + kb * 16 + nm] = c[r] * 0.03125f;  // 1/sqrt(1024)
  }

  // Row softmax; write attn to global and normalized P back into S (in place).
  float* attn_base = attn + (hb * L + (size_t)m0) * L;
  for (int r = 0; r < 16; ++r) {
    float4* Sr4 = reinterpret_cast<float4*>(&S[r * L]);
    float4 e[8];
    float m = -3.4e38f;
#pragma unroll
    for (int i = 0; i < 8; ++i) {
      float4 t = Sr4[i * 32 + lane];
      e[i] = t;
      m = fmaxf(m, fmaxf(fmaxf(t.x, t.y), fmaxf(t.z, t.w)));
    }
#pragma unroll
    for (int off = 16; off > 0; off >>= 1) m = fmaxf(m, __shfl_xor(m, off, 32));
    float s = 0.f;
#pragma unroll
    for (int i = 0; i < 8; ++i) {
      e[i].x = __expf(e[i].x - m);
      e[i].y = __expf(e[i].y - m);
      e[i].z = __expf(e[i].z - m);
      e[i].w = __expf(e[i].w - m);
      s += e[i].x + e[i].y + e[i].z + e[i].w;
    }
#pragma unroll
    for (int off = 16; off > 0; off >>= 1) s += __shfl_xor(s, off, 32);
    const float inv = 1.0f / s;
    float4* A4 = reinterpret_cast<float4*>(attn_base + (size_t)r * L);
#pragma unroll
    for (int i = 0; i < 8; ++i) {
      float4 p;
      p.x = e[i].x * inv; p.y = e[i].y * inv;
      p.z = e[i].z * inv; p.w = e[i].w * inv;
      Sr4[i * 32 + lane] = p;   // normalized probs for PV (same-lane, in place)
      A4[i * 32 + lane] = p;    // coalesced attn store
    }
  }

  // O = P @ V   (A from LDS f32 with cvt, B from V bf16 rows)
  v8f o[4] = {{}, {}, {}, {}};
  for (int kk = 0; kk < L / 32; ++kk) {
    v16bf a = load_a_f32cvt(&S[0], L, 0, kk * 32, lane);
#pragma unroll
    for (int dt = 0; dt < 4; ++dt)
      o[dt] = wmma_bf16(a, load_b_bf16(Vh, D, kk * 32, dt * 16, lane), o[dt]);
  }
#pragma unroll
  for (int dt = 0; dt < 4; ++dt)
#pragma unroll
    for (int r = 0; r < 8; ++r)
      Obuf[(r + hf * 8) * D + dt * 16 + nm] = o[dt][r];

  // out = O @ Wfc, scattered to (n, l, h*64 + d)
  v16bf ao0 = load_a_f32cvt(&Obuf[0], D, 0, 0, lane);
  v16bf ao1 = load_a_f32cvt(&Obuf[0], D, 0, 32, lane);
#pragma unroll
  for (int dt = 0; dt < 4; ++dt) {
    v8f f = {};
    f = wmma_bf16(ao0, load_b_f32cvt(Wfc, D, 0, dt * 16, lane), f);
    f = wmma_bf16(ao1, load_b_f32cvt(Wfc, D, 32, dt * 16, lane), f);
#pragma unroll
    for (int r = 0; r < 8; ++r) {
      const int M = r + hf * 8;
      out[((size_t)n * L + m0 + M) * E + h * D + dt * 16 + nm] = f[r];
    }
  }
}

extern "C" void kernel_launch(void* const* d_in, const int* in_sizes, int n_in,
                              void* d_out, int out_size, void* d_ws, size_t ws_size,
                              hipStream_t stream) {
  const float* x   = (const float*)d_in[0];
  const float* Wq  = (const float*)d_in[1];
  const float* Wk  = (const float*)d_in[2];
  const float* Wv  = (const float*)d_in[3];
  const float* Wfc = (const float*)d_in[4];

  float* out  = (float*)d_out;
  float* attn = out + (size_t)N * L * E;   // outputs concatenated: out, attn

  bf16_t* Qs = (bf16_t*)d_ws;              // 16 MB
  bf16_t* Kt = Qs + (size_t)N * H * L * D; // 16 MB
  bf16_t* Vs = Kt + (size_t)N * H * L * D; // 16 MB   (48 MB total workspace)

  proj_kernel<<<dim3(N * L * H / 16), dim3(32), 0, stream>>>(x, Wq, Wk, Wv, Qs, Kt, Vs);
  attn_kernel<<<dim3(L / 16, H, N), dim3(32), 0, stream>>>(Qs, Kt, Vs, Wfc, out, attn);
}
